// SecondOrderRecurrency_747324309607
// MI455X (gfx1250) — matrix operand order repaired and checked by reading, not compiled, not run on hardware
//
#include <hip/hip_runtime.h>
#include <hip/hip_bf16.h>

// ---------------------------------------------------------------------------
// Second-order recurrent propagation (BasicVSR++-style) for MI455X / gfx1250.
// Convs = implicit GEMM on v_wmma_f32_16x16x32_bf16 with 2x2 register
// blocking per wave (32co x 32pix); intermediates stored as bf16.
// ---------------------------------------------------------------------------

typedef __attribute__((ext_vector_type(16))) __bf16          v16bf;
typedef __attribute__((ext_vector_type(16))) unsigned short  v16u;
typedef __attribute__((ext_vector_type(8)))  unsigned short  v8u;
typedef __attribute__((ext_vector_type(8)))  float           v8f;

#define IMG_H 128
#define IMG_W 128
#define HWSZ  (IMG_H * IMG_W)
#define NCH   64
#define TSTEP 16
#define NB    2

// float -> bf16 (round to nearest even)
static __device__ __forceinline__ unsigned short f2bf(float f) {
    unsigned int u = __float_as_uint(f);
    u += 0x7FFFu + ((u >> 16) & 1u);
    return (unsigned short)(u >> 16);
}
// bf16 -> float
static __device__ __forceinline__ float bf2f(unsigned short s) {
    return __uint_as_float(((unsigned int)s) << 16);
}

static __device__ __forceinline__ void storeVal(float* p, float v)          { *p = v; }
static __device__ __forceinline__ void storeVal(unsigned short* p, float v) { *p = f2bf(v); }

static __device__ __forceinline__ v8f wmma_bf16(v16u a, v16u b, v8f c) {
    return __builtin_amdgcn_wmma_f32_16x16x32_bf16(
        false, __builtin_bit_cast(v16bf, a),
        false, __builtin_bit_cast(v16bf, b),
        (short)0, c, false, false);
}

// ---------------------------------------------------------------------------
// Weight conversion kernels (fp32 -> bf16, GEMM-friendly layouts)
// ---------------------------------------------------------------------------

// W_fc: [64][192][1][1] -> bf16 [co][192]
__global__ void cvt_wfc_kernel(const float* __restrict__ w,
                               unsigned short* __restrict__ out) {
    int i = blockIdx.x * 256 + threadIdx.x;
    if (i < 64 * 192) out[i] = f2bf(w[i]);
}

// W 3x3: [co][ci][3][3] -> bf16 [co][tap*64 + ci], tap = ky*3+kx  (K = 576)
__global__ void cvt_w3_kernel(const float* __restrict__ w,
                              unsigned short* __restrict__ out) {
    int i = blockIdx.x * 256 + threadIdx.x;
    if (i < 64 * 64 * 9) {
        int co  = i / (64 * 9);
        int r   = i % (64 * 9);
        int ci  = r / 9;
        int tap = r % 9;
        out[co * 576 + tap * 64 + ci] = f2bf(w[i]);
    }
}

// ---------------------------------------------------------------------------
// Bilinear warp (reference flow_warp semantics; zero outside border).
// OUT = float (flows) or unsigned short (bf16 feature intermediates).
// addSrc != null => out = addSrc + warp(feat, flow)  (used for f2c)
// ---------------------------------------------------------------------------
template <typename OUT>
__global__ void warp_kernel(const float* __restrict__ feat, long long featBS,
                            const float* __restrict__ flow, long long flowBS,
                            const float* __restrict__ addSrc, long long addBS,
                            OUT* __restrict__ out, long long outBS, int nch) {
    int p  = blockIdx.x * 256 + threadIdx.x;
    int nb = blockIdx.y;
    if (p >= HWSZ) return;
    feat += (long long)nb * featBS;
    flow += (long long)nb * flowBS;
    out  += (long long)nb * outBS;
    if (addSrc) addSrc += (long long)nb * addBS;

    int px = p & (IMG_W - 1);
    int py = p >> 7;
    float gx = (float)px + flow[p];
    float gy = (float)py + flow[HWSZ + p];
    float x0f = floorf(gx), y0f = floorf(gy);
    float wx = gx - x0f, wy = gy - y0f;
    int x0 = (int)x0f, y0 = (int)y0f;
    int x1 = x0 + 1, y1 = y0 + 1;
    float vx0 = (x0 >= 0 && x0 < IMG_W) ? 1.f : 0.f;
    float vx1 = (x1 >= 0 && x1 < IMG_W) ? 1.f : 0.f;
    float vy0 = (y0 >= 0 && y0 < IMG_H) ? 1.f : 0.f;
    float vy1 = (y1 >= 0 && y1 < IMG_H) ? 1.f : 0.f;
    int cx0 = min(max(x0, 0), IMG_W - 1), cx1 = min(max(x1, 0), IMG_W - 1);
    int cy0 = min(max(y0, 0), IMG_H - 1), cy1 = min(max(y1, 0), IMG_H - 1);
    float w00 = (1.f - wx) * (1.f - wy) * vx0 * vy0;
    float w01 = wx * (1.f - wy) * vx1 * vy0;
    float w10 = (1.f - wx) * wy * vx0 * vy1;
    float w11 = wx * wy * vx1 * vy1;
    int i00 = cy0 * IMG_W + cx0, i01 = cy0 * IMG_W + cx1;
    int i10 = cy1 * IMG_W + cx0, i11 = cy1 * IMG_W + cx1;

    for (int ci = 0; ci < nch; ++ci) {
        const float* f = feat + (long long)ci * HWSZ;
        float v = f[i00] * w00 + f[i01] * w01 + f[i10] * w10 + f[i11] * w11;
        if (addSrc) v += addSrc[(long long)ci * HWSZ + p];
        storeVal(&out[(long long)ci * HWSZ + p], v);
    }
}

// ---------------------------------------------------------------------------
// Fragment helpers (contiguous-k order used by this file for both A and B, wave32)
//   A (16x32 bf16): lane m = lane&15; K = kb..kb+15 contiguous,
//                   kb = chunk*32 + (lane>=16 ? 16 : 0)
//   B (32x16 bf16): lane n = lane&15; K = kb..kb+15 contiguous,
//                   kb = chunk*32 + (lane>=16 ? 16 : 0)
// ---------------------------------------------------------------------------
static __device__ __forceinline__ v16u loadA(const unsigned short* p) {
    v8u lo = *(const v8u*)p;
    v8u hi = *(const v8u*)(p + 8);
    v16u r;
    #pragma unroll
    for (int j = 0; j < 8; ++j) { r[j] = lo[j]; r[8 + j] = hi[j]; }
    return r;
}
static __device__ __forceinline__ v16u loadB(const unsigned short* p) {
    v8u lo = *(const v8u*)p;
    v8u hi = *(const v8u*)(p + 8);
    v16u r;
    #pragma unroll
    for (int j = 0; j < 8; ++j) { r[j] = lo[j]; r[8 + j] = hi[j]; }
    return r;
}

// ---------------------------------------------------------------------------
// 1x1 conv over [a2|a1|x] concat: GEMM K=192, full 128-pixel row per block.
// 8 waves: (coGroup = wv&1)*32 co, (pixGroup = wv>>1)*32 pixels; each wave
// computes 2x2 subtiles of 16x16 with 4 accumulators.
// a2,a1 are bf16; x is fp32 (converted while staging). Output bf16.
// ---------------------------------------------------------------------------
__global__ __launch_bounds__(256)
void convfc_wmma_kernel(const unsigned short* __restrict__ Wb,   // [64][192] bf16
                        const float* __restrict__ bias,
                        const unsigned short* __restrict__ a2,
                        const unsigned short* __restrict__ a1,
                        const float* __restrict__ x, long long xBS,
                        unsigned short* __restrict__ out) {
    __shared__ __align__(16) unsigned short tile[128][208];  // [pixel][k]

    int nb = blockIdx.z;
    int h0 = blockIdx.y;
    int pbase = h0 * IMG_W;

    const unsigned short* a2p = a2 + (long long)nb * NCH * HWSZ;
    const unsigned short* a1p = a1 + (long long)nb * NCH * HWSZ;
    const float*          xp  = x  + (long long)nb * xBS;

    // Stage 128 pixels x 192 channels, channel-contiguous per pixel.
    for (int idx = threadIdx.x; idx < 128 * 192; idx += 256) {
        int k = idx >> 7;          // 0..191 concat channel
        int c = idx & 127;         // pixel in row
        unsigned short v;
        if (k < 64)       v = a2p[(long long)k * HWSZ + pbase + c];
        else if (k < 128) v = a1p[(long long)(k - 64) * HWSZ + pbase + c];
        else              v = f2bf(xp[(long long)(k - 128) * HWSZ + pbase + c]);
        tile[c][k] = v;
    }
    __syncthreads();

    int wv       = threadIdx.x >> 5;
    int lane     = threadIdx.x & 31;
    int coBase0  = (wv & 1) * 32;
    int pixBase0 = (wv >> 1) * 32;
    int lm       = lane & 15;
    bool hiLane  = lane >= 16;

    v8f acc[2][2] = {};
    #pragma unroll
    for (int ch = 0; ch < 6; ++ch) {
        int kbA = ch * 32 + (hiLane ? 16 : 0);
        int kbB = ch * 32 + (hiLane ? 16 : 0);
        v16u Af[2], Bf[2];
        #pragma unroll
        for (int ct = 0; ct < 2; ++ct)
            Af[ct] = loadA(Wb + (coBase0 + ct * 16 + lm) * 192 + kbA);
        #pragma unroll
        for (int pt = 0; pt < 2; ++pt)
            Bf[pt] = loadB(&tile[pixBase0 + pt * 16 + lm][kbB]);
        #pragma unroll
        for (int ct = 0; ct < 2; ++ct)
            #pragma unroll
            for (int pt = 0; pt < 2; ++pt)
                acc[ct][pt] = wmma_bf16(Af[ct], Bf[pt], acc[ct][pt]);
    }

    unsigned short* o = out + (long long)nb * NCH * HWSZ;
    #pragma unroll
    for (int ct = 0; ct < 2; ++ct) {
        #pragma unroll
        for (int pt = 0; pt < 2; ++pt) {
            int pix = pbase + pixBase0 + pt * 16 + lm;
            #pragma unroll
            for (int r = 0; r < 8; ++r) {
                int co = coBase0 + ct * 16 + r + (hiLane ? 8 : 0);
                o[(long long)co * HWSZ + pix] = f2bf(acc[ct][pt][r] + bias[co]);
            }
        }
    }
}

// ---------------------------------------------------------------------------
// 3x3 SAME conv 64->64 implicit GEMM, K = 9*64 = 576 (18 chunks of 32),
// full 128-pixel row per block, 2x2 register blocking per wave.
// Input bf16 [64][HW].  MODE 0: bias+leaky_relu -> bf16 out
//                       MODE 1: bias+residual(x fp32) -> fp32 out
// ---------------------------------------------------------------------------
template <int MODE, typename OUT>
__global__ __launch_bounds__(256)
void conv3_wmma_kernel(const unsigned short* __restrict__ Wb,   // [64][576] bf16
                       const float* __restrict__ bias,
                       const unsigned short* __restrict__ in, long long inBS,
                       const float* __restrict__ res, long long resBS,
                       OUT* __restrict__ out, long long outBS) {
    __shared__ __align__(16) unsigned short tile[3][130][72];  // [row][col][ci]

    int nb = blockIdx.z;
    int h0 = blockIdx.y;
    const unsigned short* src = in + (long long)nb * inBS;

    // Stage 3 rows x 130 cols x 64 ch halo (zero-padded borders).
    for (int idx = threadIdx.x; idx < 3 * 130 * 64; idx += 256) {
        int ci = idx / 390;
        int r  = (idx % 390) / 130;
        int c  = idx % 130;
        int gy = h0 - 1 + r;
        int gx = c - 1;
        unsigned short v = 0;
        if (gy >= 0 && gy < IMG_H && gx >= 0 && gx < IMG_W)
            v = src[(long long)ci * HWSZ + gy * IMG_W + gx];
        tile[r][c][ci] = v;
    }
    __syncthreads();

    int wv       = threadIdx.x >> 5;
    int lane     = threadIdx.x & 31;
    int coBase0  = (wv & 1) * 32;
    int pixBase0 = (wv >> 1) * 32;
    int lm       = lane & 15;
    bool hiLane  = lane >= 16;

    v8f acc[2][2] = {};
    #pragma unroll
    for (int ch = 0; ch < 18; ++ch) {
        int tap = ch >> 1;
        int ky  = tap / 3;
        int kx  = tap % 3;
        int kbA = ch * 32 + (hiLane ? 16 : 0);
        int ciB = (ch & 1) * 32 + (hiLane ? 16 : 0);

        v16u Af[2], Bf[2];
        #pragma unroll
        for (int ct = 0; ct < 2; ++ct)
            Af[ct] = loadA(Wb + (coBase0 + ct * 16 + lm) * 576 + kbA);
        #pragma unroll
        for (int pt = 0; pt < 2; ++pt)
            Bf[pt] = loadB(&tile[ky][pixBase0 + pt * 16 + lm + kx][ciB]);
        #pragma unroll
        for (int ct = 0; ct < 2; ++ct)
            #pragma unroll
            for (int pt = 0; pt < 2; ++pt)
                acc[ct][pt] = wmma_bf16(Af[ct], Bf[pt], acc[ct][pt]);
    }

    OUT* o = out + (long long)nb * outBS;
    const float* rp = (MODE == 1) ? (res + (long long)nb * resBS) : nullptr;
    int pbase = h0 * IMG_W;
    #pragma unroll
    for (int ct = 0; ct < 2; ++ct) {
        #pragma unroll
        for (int pt = 0; pt < 2; ++pt) {
            int pix = pbase + pixBase0 + pt * 16 + lm;
            #pragma unroll
            for (int r = 0; r < 8; ++r) {
                int co  = coBase0 + ct * 16 + r + (hiLane ? 8 : 0);
                float v = acc[ct][pt][r] + bias[co];
                if (MODE == 0) v = (v > 0.f) ? v : 0.1f * v;          // leaky relu
                if (MODE == 1) v += rp[(long long)co * HWSZ + pix];   // + x
                storeVal(&o[(long long)co * HWSZ + pix], v);
            }
        }
    }
}

// ---------------------------------------------------------------------------
// Host orchestration: fully unrolled 16-step scan, pointer-only recurrence.
// ---------------------------------------------------------------------------
extern "C" void kernel_launch(void* const* d_in, const int* in_sizes, int n_in,
                              void* d_out, int out_size, void* d_ws, size_t ws_size,
                              hipStream_t stream) {
    const float* curr  = (const float*)d_in[0];  // [2][16][64][128][128]
    const float* flows = (const float*)d_in[1];  // [2][16][2][128][128]
    const float* Wfc   = (const float*)d_in[2];
    const float* bfc   = (const float*)d_in[3];
    const float* W1    = (const float*)d_in[4];
    const float* b1    = (const float*)d_in[5];
    const float* W2    = (const float*)d_in[6];
    const float* b2    = (const float*)d_in[7];
    float* out = (float*)d_out;                  // [2][16][64][128][128]

    const long long HWl    = HWSZ;
    const long long featBS = (long long)TSTEP * NCH * HWl;   // batch stride in curr/out
    const long long flowBS = (long long)TSTEP * 2 * HWl;     // batch stride in flows
    const long long wsFeat = (long long)NCH * HWl;           // per-batch ws feature stride

    char* ws = (char*)d_ws;
    size_t off = 0;
    auto wsAlloc = [&](size_t bytes) -> void* {
        void* p = ws + off;
        off = (off + bytes + 255) & ~(size_t)255;
        return p;
    };
    unsigned short* a1b    = (unsigned short*)wsAlloc((size_t)NB * NCH * HWl * 2);
    unsigned short* a2b    = (unsigned short*)wsAlloc((size_t)NB * NCH * HWl * 2);
    unsigned short* infeat = (unsigned short*)wsAlloc((size_t)NB * NCH * HWl * 2);
    unsigned short* h1     = (unsigned short*)wsAlloc((size_t)NB * NCH * HWl * 2);
    float* f2c   = (float*)wsAlloc((size_t)NB * 2 * HWl * 4);
    float* zflow = (float*)wsAlloc((size_t)NB * 2 * HWl * 4);
    unsigned short* Wfcb = (unsigned short*)wsAlloc(64 * 192 * 2);
    unsigned short* W1b  = (unsigned short*)wsAlloc(64 * 576 * 2);
    unsigned short* W2b  = (unsigned short*)wsAlloc(64 * 576 * 2);

    hipMemsetAsync(zflow, 0, (size_t)NB * 2 * HWl * 4, stream);
    cvt_wfc_kernel<<<(64 * 192 + 255) / 256, 256, 0, stream>>>(Wfc, Wfcb);
    cvt_w3_kernel <<<(64 * 576 + 255) / 256, 256, 0, stream>>>(W1, W1b);
    cvt_w3_kernel <<<(64 * 576 + 255) / 256, 256, 0, stream>>>(W2, W2b);

    dim3 gw(HWSZ / 256, NB);   // warp kernels
    dim3 gc(1, IMG_H, NB);     // conv kernels (full 128-pixel row per block)

    for (int t = 0; t < TSTEP; ++t) {
        const float* x = curr + (long long)t * NCH * HWl;

        const float* fl1; long long f1BS;
        if (t >= 1) { fl1 = flows + (long long)(t - 1) * 2 * HWl; f1BS = flowBS; }
        else        { fl1 = zflow;                                 f1BS = 2 * HWl; }
        const float* fl2; long long f2BS;
        if (t >= 2) { fl2 = flows + (long long)(t - 2) * 2 * HWl; f2BS = flowBS; }
        else        { fl2 = zflow;                                 f2BS = 2 * HWl; }
        const float* y1 = (t >= 1) ? (out + (long long)(t - 1) * NCH * HWl) : curr;
        const float* y2 = (t >= 2) ? (out + (long long)(t - 2) * NCH * HWl) : curr;

        // a1 = warp(y1, f1)                              (bf16 out)
        warp_kernel<unsigned short><<<gw, 256, 0, stream>>>(
            y1, featBS, fl1, f1BS, nullptr, 0, a1b, wsFeat, NCH);
        // f2c = f1 + warp(f2, f1)                        (fp32 flow out)
        warp_kernel<float><<<gw, 256, 0, stream>>>(
            fl2, f2BS, fl1, f1BS, fl1, f1BS, f2c, 2 * HWl, 2);
        // a2 = warp(y2, f2c)                             (bf16 out)
        warp_kernel<unsigned short><<<gw, 256, 0, stream>>>(
            y2, featBS, f2c, 2 * HWl, nullptr, 0, a2b, wsFeat, NCH);
        // in_feat = W_fc . [a2|a1|x] + b_fc              (bf16 out)
        convfc_wmma_kernel<<<gc, 256, 0, stream>>>(Wfcb, bfc, a2b, a1b,
                                                   x, featBS, infeat);
        // h1 = leaky_relu(W1 * in_feat + b1)             (bf16 out)
        conv3_wmma_kernel<0, unsigned short><<<gc, 256, 0, stream>>>(
            W1b, b1, infeat, wsFeat, nullptr, 0, h1, wsFeat);
        // out_t = W2 * h1 + b2 + x                       (fp32 out)
        conv3_wmma_kernel<1, float><<<gc, 256, 0, stream>>>(
            W2b, b2, h1, wsFeat, x, featBS,
            out + (long long)t * NCH * HWl, featBS);
    }
}
